// Attention_1537598292135
// MI455X (gfx1250) — compile-verified, matrix operand order checked by reading, not run on hardware
//
#include <hip/hip_runtime.h>
#include <hip/hip_bf16.h>

// ---------------------------------------------------------------------------
// MHA forward for MI455X (gfx1250, wave32, WMMA 16x16x32 f16 -> f32).
//   K1: qkv = x @ Wqkv + bqkv  -> Q(scaled)/K as [B,H,N,D] f16, V^T [B,H,D,N] f16
//   K2: flash attention; K/V chunks staged into LDS by the Tensor Data Mover
//       (double-buffered tensor_load_to_lds with row padding for conflict-free
//       b128 fragment reads), online softmax with the S^T packing trick.
//   K3: out = ctx @ Wproj + bproj (f32 output)
// ---------------------------------------------------------------------------

typedef _Float16 v16h __attribute__((ext_vector_type(16)));
typedef _Float16 h8   __attribute__((ext_vector_type(8)));
typedef float    v8f  __attribute__((ext_vector_type(8)));
typedef float    f4   __attribute__((ext_vector_type(4)));
typedef int      i4   __attribute__((ext_vector_type(4)));

#define POS_INF __builtin_inff()

#if defined(__HIP_DEVICE_COMPILE__) && defined(__has_builtin)
#if __has_builtin(__builtin_amdgcn_tensor_load_to_lds) && \
    __has_builtin(__builtin_amdgcn_s_wait_tensorcnt)
#define HAVE_TDM 1
#endif
#endif
#ifndef HAVE_TDM
#define HAVE_TDM 0
#endif

__device__ __forceinline__ v8f wmma16(v16h a, v16h b, v8f c) {
  // (neg_a, A, neg_b, B, c_mod, C, reuse_a, reuse_b)
  return __builtin_amdgcn_wmma_f32_16x16x32_f16(false, a, false, b, (short)0, c,
                                                false, false);
}

// Fragment layout (wave32, 16-bit A/B, K=32): lane group g = lane/16.
//   half i in [0,8):  K = g*8 + i        (first b128)
//   half i in [8,16): K = 16 + g*8 + i-8 (second b128)
__device__ __forceinline__ v16h frag_h(const _Float16* p0, const _Float16* p1) {
  h8 a = *(const h8*)p0;
  h8 b = *(const h8*)p1;
  return __builtin_shufflevector(a, b, 0, 1, 2, 3, 4, 5, 6, 7, 8, 9, 10, 11, 12,
                                 13, 14, 15);
}

__device__ __forceinline__ v16h frag_f32(const float* p0, const float* p1) {
  f4 a0 = *(const f4*)p0, a1 = *(const f4*)(p0 + 4);
  f4 b0 = *(const f4*)p1, b1 = *(const f4*)(p1 + 4);
  v16h r;
#pragma unroll
  for (int i = 0; i < 4; ++i) {
    r[i]      = (_Float16)a0[i];
    r[4 + i]  = (_Float16)a1[i];
    r[8 + i]  = (_Float16)b0[i];
    r[12 + i] = (_Float16)b1[i];
  }
  return r;
}

#if HAVE_TDM
typedef unsigned int u32x4 __attribute__((ext_vector_type(4)));
typedef int          i32x8 __attribute__((ext_vector_type(8)));
typedef int          i32x4 __attribute__((ext_vector_type(4)));

// Issue one 2-D TDM tile load (f16 elements) with per-row LDS padding.
//   D# group0: count=1, lds_addr, global tile addr, type=2 (ISA 8.3)
//   D# group1: data_size=2B, pad_enable, pad_interval/amount, tensor dims,
//              tile dims, dim0 stride (ISA 8.4)
// Groups 2/3 are zero (2-D tensor -> unused), matching the 6-arg builtin of
// this toolchain: (u32x4, i32x8, i32x4, i32x4, i32x8, i32 cpol).
__device__ __forceinline__ void tdm_load_2d_f16(
    const void* gptr, unsigned lds_off, unsigned dim0, unsigned dim1,
    unsigned stride0, unsigned tile0, unsigned tile1, unsigned pad_interval,
    unsigned pad_amount) {
  unsigned long long ga = (unsigned long long)gptr;
  u32x4 g0;
  g0[0] = 1u;                                         // count=1 (valid user D#)
  g0[1] = lds_off;                                    // lds_addr (bytes)
  g0[2] = (unsigned)(ga & 0xffffffffu);               // global_addr[31:0]
  g0[3] = (unsigned)((ga >> 32) & 0x1ffffffu) | (2u << 30);  // [56:32] | type=2
  i32x8 g1;
  g1[0] = (int)((1u << 16) |                          // data_size = 2 bytes
                (1u << 20) |                          // pad_enable
                (pad_interval << 22) | (pad_amount << 25));
  g1[1] = (int)((dim0 & 0xffffu) << 16);              // tensor_dim0[15:0]
  g1[2] = (int)(((dim0 >> 16) & 0xffffu) | ((dim1 & 0xffffu) << 16));
  g1[3] = (int)(((dim1 >> 16) & 0xffffu) | ((tile0 & 0xffffu) << 16));
  g1[4] = (int)(tile1 & 0xffffu);                     // tile_dim1 (tile_dim2=0)
  g1[5] = (int)stride0;                               // tensor_dim0_stride[31:0]
  g1[6] = 0;
  g1[7] = 0;
  i32x4 z4 = {};
  i32x8 z8 = {};
  __builtin_amdgcn_tensor_load_to_lds(g0, g1, z4, z4, z8, 0);
}
#endif

// ---------------------------------------------------------------------------
// K1: QKV GEMM.  M=8192 (B*N), K=1024, Ncols=3072.  Block = 64x64 tile,
// 8 waves; each wave -> two 16x16 C tiles.  A fragments straight from global;
// Wqkv staged transposed in LDS (pad stride 40 -> conflict-free b128).
// ---------------------------------------------------------------------------
__global__ __launch_bounds__(256) void qkv_wmma_kernel(
    const float* __restrict__ x, const float* __restrict__ Wqkv,
    const float* __restrict__ bqkv, _Float16* __restrict__ qbuf,
    _Float16* __restrict__ kbuf, _Float16* __restrict__ vtbuf) {
  __shared__ _Float16 Blds[64][40];  // [col][k]
  const int tid  = threadIdx.x;
  const int wave = tid >> 5, lane = tid & 31;
  const int ln = lane & 15, g = lane >> 4;
  const int m_blk = blockIdx.x * 64;
  const int c_blk = blockIdx.y * 64;
  const int wm = (wave >> 2) * 32;
  const int wn = (wave & 3) * 16;

  v8f acc0 = {}, acc1 = {};
  const float* arow0 = x + (size_t)(m_blk + wm + ln) * 1024;
  const float* arow1 = arow0 + (size_t)16 * 1024;

  const int skk = tid >> 3;       // 0..31 : staged k row
  const int scc = (tid & 7) * 8;  // 0..56 : staged col base

  for (int kb = 0; kb < 1024; kb += 32) {
    const float* wp = Wqkv + (size_t)(kb + skk) * 3072 + c_blk + scc;
    f4 w0 = *(const f4*)wp;
    f4 w1 = *(const f4*)(wp + 4);
    __syncthreads();
#pragma unroll
    for (int j = 0; j < 4; ++j) {
      Blds[scc + j][skk]     = (_Float16)w0[j];
      Blds[scc + 4 + j][skk] = (_Float16)w1[j];
    }
    __syncthreads();

    v16h a0 = frag_f32(arow0 + kb + g * 8, arow0 + kb + 16 + g * 8);
    v16h a1 = frag_f32(arow1 + kb + g * 8, arow1 + kb + 16 + g * 8);
    v16h bf = frag_h(&Blds[wn + ln][g * 8], &Blds[wn + ln][16 + g * 8]);
    acc0 = wmma16(a0, bf, acc0);
    acc1 = wmma16(a1, bf, acc1);
  }

  const int c = c_blk + wn + ln;  // uniform t/h within the 16 lanes
  const float bias = bqkv[c];
  const int t  = c >> 10;
  const int hh = (c >> 6) & 15;
  const int dd = c & 63;
  const float qs = (t == 0) ? 0.125f : 1.0f;  // 1/sqrt(64) folded into Q
#pragma unroll
  for (int half = 0; half < 2; ++half) {
    v8f acc = half ? acc1 : acc0;
#pragma unroll
    for (int r = 0; r < 8; ++r) {
      int mrow = m_blk + wm + half * 16 + r + 8 * g;  // C-layout: M = r + 8g
      int bb = mrow >> 11, nn = mrow & 2047;
      _Float16 hv = (_Float16)((acc[r] + bias) * qs);
      if (t == 0)
        qbuf[(((size_t)(bb * 16 + hh)) * 2048 + nn) * 64 + dd] = hv;
      else if (t == 1)
        kbuf[(((size_t)(bb * 16 + hh)) * 2048 + nn) * 64 + dd] = hv;
      else
        vtbuf[(((size_t)(bb * 16 + hh)) * 64 + dd) * 2048 + nn] = hv;
    }
  }
}

// ---------------------------------------------------------------------------
// K2: flash attention.  Block = 8 waves x 16 queries = 128 queries of one
// (b,h); all waves march over the same 32-key chunks.  TDM double-buffers the
// K chunk (32x64, padded rows of 72 halves) and V^T chunk (64x32, padded rows
// of 40 halves) into LDS; wave 0 issues, TENSORcnt + barrier publishes.
// S^T = K*Q^T so the C-layout converts in-lane into the P*V A-fragment.
// ---------------------------------------------------------------------------
__global__ __launch_bounds__(256) void flash_attn_kernel(
    const _Float16* __restrict__ qbuf, const _Float16* __restrict__ kbuf,
    const _Float16* __restrict__ vtbuf, const int* __restrict__ mask,
    _Float16* __restrict__ ctx) {
  const int bh = blockIdx.x;  // 0..63
  const int b = bh >> 4, h = bh & 15;
  const int wave = threadIdx.x >> 5, lane = threadIdx.x & 31;
  const int ln = lane & 15, g = lane >> 4;
  const int q0 = blockIdx.y * 128 + wave * 16;

  // Q fragments (B operand: column = query = lane%16), reused for all keys.
  const _Float16* Qp = qbuf + ((size_t)bh * 2048 + q0 + ln) * 64;
  const v16h qlo = frag_h(Qp + g * 8, Qp + 16 + g * 8);       // d 0..31
  const v16h qhi = frag_h(Qp + 32 + g * 8, Qp + 48 + g * 8);  // d 32..63

  const _Float16* Kb = kbuf + (size_t)bh * 2048 * 64;
  const _Float16* Vb = vtbuf + (size_t)bh * 64 * 2048;
  const int* Mrow = mask + ((size_t)b * 2048 + q0 + ln) * 2048;

  v8f o0 = {}, o1 = {}, o2 = {}, o3 = {};  // O tiles, d = t*16 + lane%16
  float m_run = -POS_INF;
  float l_run = 0.0f;

#if HAVE_TDM
  __shared__ _Float16 Kl[2][32 * 72];  // 32 key rows, 64+8 halves (144B) each
  __shared__ _Float16 Vl[2][64 * 40];  // 64 d rows, 32+8 halves (80B) each
  const bool issuer = (wave == 0);
  if (issuer) {
    // K tile: tensor [keys=2048][d=64] row-major; tile 32x64; pad 128B->+16B.
    tdm_load_2d_f16(Kb, (unsigned)(size_t)&Kl[0][0], 64u, 2048u, 64u, 64u, 32u,
                    4u, 3u);
    // V^T tile: tensor [d=64][keys=2048] row-major; tile 64x32; pad 64B->+16B.
    tdm_load_2d_f16(Vb, (unsigned)(size_t)&Vl[0][0], 2048u, 64u, 2048u, 32u,
                    64u, 3u, 3u);
  }
#endif

  for (int it = 0; it < 64; ++it) {
    const int kc = it << 5;
#if HAVE_TDM
    const int cur = it & 1;
    if (issuer) {
      if (it + 1 < 64) {
        const int kn = kc + 32;
        tdm_load_2d_f16(Kb + (size_t)kn * 64, (unsigned)(size_t)&Kl[cur ^ 1][0],
                        64u, 2048u, 64u, 64u, 32u, 4u, 3u);
        tdm_load_2d_f16(Vb + kn, (unsigned)(size_t)&Vl[cur ^ 1][0], 2048u, 64u,
                        2048u, 32u, 64u, 3u, 3u);
        // <=2 outstanding: only the prefetched pair; current pair complete.
        __builtin_amdgcn_s_wait_tensorcnt((short)2);
      } else {
        __builtin_amdgcn_s_wait_tensorcnt((short)0);
      }
    }
    __syncthreads();
#endif

    // ---- acquire K / V fragments for this chunk
    v16h k0lo, k0hi, k1lo, k1hi, vf0, vf1, vf2, vf3;
#if HAVE_TDM
    {
      const _Float16* r0 = &Kl[cur][(size_t)ln * 72];
      const _Float16* r1 = r0 + 16 * 72;
      k0lo = frag_h(r0 + g * 8, r0 + 16 + g * 8);
      k0hi = frag_h(r0 + 32 + g * 8, r0 + 48 + g * 8);
      k1lo = frag_h(r1 + g * 8, r1 + 16 + g * 8);
      k1hi = frag_h(r1 + 32 + g * 8, r1 + 48 + g * 8);
      const _Float16* v0 = &Vl[cur][(size_t)ln * 40];
      vf0 = frag_h(v0 + g * 8, v0 + 16 + g * 8);
      const _Float16* v1 = v0 + 16 * 40;
      vf1 = frag_h(v1 + g * 8, v1 + 16 + g * 8);
      const _Float16* v2 = v1 + 16 * 40;
      vf2 = frag_h(v2 + g * 8, v2 + 16 + g * 8);
      const _Float16* v3 = v2 + 16 * 40;
      vf3 = frag_h(v3 + g * 8, v3 + 16 + g * 8);
    }
#else
    {
      const _Float16* r0 = Kb + (size_t)(kc + ln) * 64;
      const _Float16* r1 = r0 + 16 * 64;
      k0lo = frag_h(r0 + g * 8, r0 + 16 + g * 8);
      k0hi = frag_h(r0 + 32 + g * 8, r0 + 48 + g * 8);
      k1lo = frag_h(r1 + g * 8, r1 + 16 + g * 8);
      k1hi = frag_h(r1 + 32 + g * 8, r1 + 48 + g * 8);
      const _Float16* v0 = Vb + (size_t)ln * 2048 + kc;
      vf0 = frag_h(v0 + g * 8, v0 + 16 + g * 8);
      const _Float16* v1 = v0 + (size_t)16 * 2048;
      vf1 = frag_h(v1 + g * 8, v1 + 16 + g * 8);
      const _Float16* v2 = v1 + (size_t)16 * 2048;
      vf2 = frag_h(v2 + g * 8, v2 + 16 + g * 8);
      const _Float16* v3 = v2 + (size_t)16 * 2048;
      vf3 = frag_h(v3 + g * 8, v3 + 16 + g * 8);
    }
#endif

    // ---- S^T tiles: rows = keys, cols = queries
    v8f s0 = {}, s1 = {};
    s0 = wmma16(k0lo, qlo, s0);
    s0 = wmma16(k0hi, qhi, s0);
    s1 = wmma16(k1lo, qlo, s1);
    s1 = wmma16(k1hi, qhi, s1);

    // ---- mask: element r of s0 <-> key kc + 8g + r ; s1 <-> +16
    i4 m0a = *(const i4*)(Mrow + kc + g * 8);
    i4 m0b = *(const i4*)(Mrow + kc + g * 8 + 4);
    i4 m1a = *(const i4*)(Mrow + kc + 16 + g * 8);
    i4 m1b = *(const i4*)(Mrow + kc + 16 + g * 8 + 4);
#pragma unroll
    for (int r = 0; r < 4; ++r) {
      if (m0a[r]) s0[r] = -POS_INF;
      if (m0b[r]) s0[4 + r] = -POS_INF;
      if (m1a[r]) s1[r] = -POS_INF;
      if (m1b[r]) s1[4 + r] = -POS_INF;
    }

    // ---- online softmax (per query = lane%16; partner lane = xor 16)
    float cm = -POS_INF;
#pragma unroll
    for (int r = 0; r < 8; ++r) cm = fmaxf(cm, fmaxf(s0[r], s1[r]));
    cm = fmaxf(cm, __shfl_xor(cm, 16));
    const float mn = fmaxf(m_run, cm);
    const float corr = __expf(m_run - mn);
    m_run = mn;

    v16h pa;  // P A-fragment: halves 0..7 <-> keys kc+8g+r, 8..15 <-> +16
    float lsum = 0.0f;
#pragma unroll
    for (int r = 0; r < 8; ++r) {
      float p0 = __expf(s0[r] - mn);
      float p1 = __expf(s1[r] - mn);
      lsum += p0 + p1;
      pa[r] = (_Float16)p0;
      pa[8 + r] = (_Float16)p1;
    }
    lsum += __shfl_xor(lsum, 16);
    l_run = l_run * corr + lsum;

    // ---- rescale O (row-space: query = r + 8g -> factor from lane r+8g)
    float cf[8];
#pragma unroll
    for (int r = 0; r < 8; ++r) cf[r] = __shfl(corr, g * 8 + r);
#pragma unroll
    for (int r = 0; r < 8; ++r) {
      o0[r] *= cf[r]; o1[r] *= cf[r]; o2[r] *= cf[r]; o3[r] *= cf[r];
    }

    // ---- O += P * V
    o0 = wmma16(pa, vf0, o0);
    o1 = wmma16(pa, vf1, o1);
    o2 = wmma16(pa, vf2, o2);
    o3 = wmma16(pa, vf3, o3);

#if HAVE_TDM
    __syncthreads();  // all reads of buf[cur] done before it is re-staged
#endif
  }

  const float linv = 1.0f / l_run;
  float lf[8];
#pragma unroll
  for (int r = 0; r < 8; ++r) lf[r] = __shfl(linv, g * 8 + r);
#pragma unroll
  for (int r = 0; r < 8; ++r) {
    const size_t q = (size_t)q0 + r + 8 * g;
    _Float16* cp = ctx + ((size_t)b * 2048 + q) * 1024 + h * 64 + ln;
    cp[0]  = (_Float16)(o0[r] * lf[r]);
    cp[16] = (_Float16)(o1[r] * lf[r]);
    cp[32] = (_Float16)(o2[r] * lf[r]);
    cp[48] = (_Float16)(o3[r] * lf[r]);
  }
}

// ---------------------------------------------------------------------------
// K3: projection GEMM.  M=8192, K=1024, Ncols=1024.  ctx is f16 -> direct
// global A fragments; Wproj transposed through LDS like K1.
// ---------------------------------------------------------------------------
__global__ __launch_bounds__(256) void proj_wmma_kernel(
    const _Float16* __restrict__ ctx, const float* __restrict__ Wproj,
    const float* __restrict__ bproj, float* __restrict__ out) {
  __shared__ _Float16 Blds[64][40];
  const int tid  = threadIdx.x;
  const int wave = tid >> 5, lane = tid & 31;
  const int ln = lane & 15, g = lane >> 4;
  const int m_blk = blockIdx.x * 64;
  const int c_blk = blockIdx.y * 64;
  const int wm = (wave >> 2) * 32;
  const int wn = (wave & 3) * 16;

  v8f acc0 = {}, acc1 = {};
  const _Float16* arow0 = ctx + (size_t)(m_blk + wm + ln) * 1024;
  const _Float16* arow1 = arow0 + (size_t)16 * 1024;

  const int skk = tid >> 3;
  const int scc = (tid & 7) * 8;

  for (int kb = 0; kb < 1024; kb += 32) {
    const float* wp = Wproj + (size_t)(kb + skk) * 1024 + c_blk + scc;
    f4 w0 = *(const f4*)wp;
    f4 w1 = *(const f4*)(wp + 4);
    __syncthreads();
#pragma unroll
    for (int j = 0; j < 4; ++j) {
      Blds[scc + j][skk]     = (_Float16)w0[j];
      Blds[scc + 4 + j][skk] = (_Float16)w1[j];
    }
    __syncthreads();

    v16h a0 = frag_h(arow0 + kb + g * 8, arow0 + kb + 16 + g * 8);
    v16h a1 = frag_h(arow1 + kb + g * 8, arow1 + kb + 16 + g * 8);
    v16h bf = frag_h(&Blds[wn + ln][g * 8], &Blds[wn + ln][16 + g * 8]);
    acc0 = wmma16(a0, bf, acc0);
    acc1 = wmma16(a1, bf, acc1);
  }

  const int c = c_blk + wn + ln;
  const float bias = bproj[c];
#pragma unroll
  for (int r = 0; r < 8; ++r) {
    int m = m_blk + wm + r + 8 * g;
    out[(size_t)m * 1024 + c]        = acc0[r] + bias;
    out[(size_t)(m + 16) * 1024 + c] = acc1[r] + bias;
  }
}

// ---------------------------------------------------------------------------
extern "C" void kernel_launch(void* const* d_in, const int* in_sizes, int n_in,
                              void* d_out, int out_size, void* d_ws,
                              size_t ws_size, hipStream_t stream) {
  (void)in_sizes; (void)n_in; (void)out_size; (void)ws_size;
  const float* x     = (const float*)d_in[0];
  const int*   mask  = (const int*)d_in[1];
  const float* Wqkv  = (const float*)d_in[2];
  const float* bqkv  = (const float*)d_in[3];
  const float* Wproj = (const float*)d_in[4];
  const float* bproj = (const float*)d_in[5];
  float* out = (float*)d_out;

  // Workspace: Q, K, V^T, ctx as f16 -> 4 * 16 MB = 64 MB.
  const size_t nqkv = (size_t)4 * 16 * 2048 * 64;
  _Float16* qbuf  = (_Float16*)d_ws;
  _Float16* kbuf  = qbuf + nqkv;
  _Float16* vtbuf = kbuf + nqkv;
  _Float16* ctx   = vtbuf + nqkv;

  qkv_wmma_kernel<<<dim3(128, 48), 256, 0, stream>>>(x, Wqkv, bqkv, qbuf, kbuf,
                                                     vtbuf);
  flash_attn_kernel<<<dim3(64, 16), 256, 0, stream>>>(qbuf, kbuf, vtbuf, mask,
                                                      ctx);
  proj_wmma_kernel<<<dim3(128, 16), 256, 0, stream>>>(ctx, Wproj, bproj, out);
}
